// MultiHeadAttentionLayer_9560597201454
// MI455X (gfx1250) — matrix operand order repaired and checked by reading, not compiled, not run on hardware
//
#include <hip/hip_runtime.h>
#include <hip/hip_bf16.h>
#include <stdint.h>

#define SEQ 4096
#define HID 768
#define NHEAD 12
#define HDIM 64

typedef __bf16 bf16_t;
typedef __attribute__((ext_vector_type(16))) __bf16 v16bf;
typedef __attribute__((ext_vector_type(8)))  float  v8f;
typedef __attribute__((ext_vector_type(4)))  int    v4i;

union FragBF {
    v16bf    v;
    uint32_t u[8];
    uint4    q[2];
};

// pack two fp32 -> dword of two bf16 (round-half-up) in 3 VALU ops:
// two adds + one v_perm_b32 (bytes 0-3 from S1=lo, 4-7 from S0=hi)
__device__ __forceinline__ uint32_t pack_bf16x2(float lo, float hi) {
    uint32_t ul = __builtin_bit_cast(uint32_t, lo) + 0x8000u;
    uint32_t uh = __builtin_bit_cast(uint32_t, hi) + 0x8000u;
    return __builtin_amdgcn_perm(uh, ul, 0x07060302u);
}

__device__ __forceinline__ bf16_t f2bf(float f) {
    uint32_t u = __builtin_bit_cast(uint32_t, f) + 0x8000u;
    return __builtin_bit_cast(bf16_t, (uint16_t)(u >> 16));
}

#if __has_builtin(__builtin_amdgcn_global_load_async_to_lds_b128) && \
    __has_builtin(__builtin_amdgcn_s_wait_asynccnt)
#define HAVE_ASYNC_LDS 1
// builtin takes v4i pointers; generic LDS pointer low 32 bits == LDS offset
#define AS1(p) ((__attribute__((address_space(1))) v4i*)(uintptr_t)(p))
#define AS3(p) ((__attribute__((address_space(3))) v4i*)(uint32_t)(uintptr_t)(p))
#endif

// ---------------------------------------------------------------------------
// Kernel 0: one-shot fp32 -> bf16 conversion (8 elements / thread).
// ---------------------------------------------------------------------------
__global__ __launch_bounds__(256)
void to_bf16(const float* __restrict__ src, bf16_t* __restrict__ dst, int n)
{
    int i = (blockIdx.x * 256 + threadIdx.x) * 8;
    if (i >= n) return;
    const float4* p = (const float4*)(src + i);
    float4 f0 = p[0], f1 = p[1];
    uint4 o;
    o.x = pack_bf16x2(f0.x, f0.y);
    o.y = pack_bf16x2(f0.z, f0.w);
    o.z = pack_bf16x2(f1.x, f1.y);
    o.w = pack_bf16x2(f1.z, f1.w);
    *(uint4*)(dst + i) = o;
}

// ---------------------------------------------------------------------------
// Kernel 1: fused Q/K/V projection on pre-converted bf16, 64x64 tile per
// wave: inner loop is 16 b128 loads + 16 WMMAs per K-step (no conversion).
//   Q, K stored head-major row-major: [h][l][d]
//   V stored head-major TRANSPOSED:   [h][d][l]
// grid = (HID/64, SEQ/64, 3)  block = 32
// ---------------------------------------------------------------------------
__global__ __launch_bounds__(32)
void qkv_proj(const bf16_t* __restrict__ Xb, const bf16_t* __restrict__ Wb,
              const float* __restrict__ bq, const float* __restrict__ bk,
              const float* __restrict__ bv,
              bf16_t* __restrict__ outQ, bf16_t* __restrict__ outK,
              bf16_t* __restrict__ outVt)
{
    const int lane = threadIdx.x;
    const int n0 = blockIdx.x * 64;   // 64-aligned -> single head per tile
    const int m0 = blockIdx.y * 64;
    const int z  = blockIdx.z;

    const bf16_t* X  = Xb + (size_t)z * SEQ * HID;
    const bf16_t* W  = Wb + (size_t)z * HID * HID;
    const float*  Bs = (z == 0) ? bq : (z == 1) ? bk : bv;

    const int half = lane >> 4;
    const int lp   = lane & 15;

    const bf16_t* xrow[4];
    const bf16_t* wrow[4];
#pragma unroll
    for (int i = 0; i < 4; ++i) {
        xrow[i] = X + (size_t)(m0 + i * 16 + lp) * HID;   // A rows
        wrow[i] = W + (size_t)(n0 + i * 16 + lp) * HID;   // B cols (W rows)
    }

    v8f acc[4][4] = {};
    for (int kc = 0; kc < HID; kc += 32) {
        FragBF a[4];
#pragma unroll
        for (int mi = 0; mi < 4; ++mi) {
            // A 16x32: 8 contiguous k at kc+8*half, 8 more at +16
            a[mi].q[0] = *(const uint4*)(xrow[mi] + kc + (half << 3));
            a[mi].q[1] = *(const uint4*)(xrow[mi] + kc + 16 + (half << 3));
        }
#pragma unroll
        for (int ni = 0; ni < 4; ++ni) {
            FragBF b;   // B 32x16: 8 contiguous k at kc+8*half, 8 more at +16
            b.q[0] = *(const uint4*)(wrow[ni] + kc + (half << 3));
            b.q[1] = *(const uint4*)(wrow[ni] + kc + 16 + (half << 3));
#pragma unroll
            for (int mi = 0; mi < 4; ++mi)
                acc[mi][ni] = __builtin_amdgcn_wmma_f32_16x16x32_bf16(
                    false, a[mi].v, false, b.v, (short)0, acc[mi][ni], false, false);
        }
    }

    const int h = n0 >> 6;   // whole tile belongs to one head
#pragma unroll
    for (int ni = 0; ni < 4; ++ni) {
        const int   d  = ni * 16 + lp;          // column within head
        const float bb = Bs[n0 + ni * 16 + lp];
        if (z == 2) {
            // V transposed [h][d][m]: 8 consecutive m -> one uint4 store
#pragma unroll
            for (int mi = 0; mi < 4; ++mi) {
                v8f& c = acc[mi][ni];
                uint4 pk;
                pk.x = pack_bf16x2(c[0] + bb, c[1] + bb);
                pk.y = pack_bf16x2(c[2] + bb, c[3] + bb);
                pk.z = pack_bf16x2(c[4] + bb, c[5] + bb);
                pk.w = pack_bf16x2(c[6] + bb, c[7] + bb);
                *(uint4*)(outVt + ((size_t)h * HDIM + d) * SEQ +
                          m0 + mi * 16 + (half << 3)) = pk;
            }
        } else {
            bf16_t* Y = (z == 0) ? outQ : outK;
#pragma unroll
            for (int mi = 0; mi < 4; ++mi)
#pragma unroll
                for (int r = 0; r < 8; ++r) {
                    int m = m0 + mi * 16 + r + (half << 3);
                    Y[((size_t)h * SEQ + m) * HDIM + d] = f2bf(acc[mi][ni][r] + bb);
                }
        }
    }
}

// ---------------------------------------------------------------------------
// Kernel 2: flash attention.  8 waves/block (128 query rows); 32-key chunks
// async-staged to LDS; fragments are ds_load_b128 pairs.
// grid = (SEQ/128, NHEAD)  block = 256
// ---------------------------------------------------------------------------
__global__ __launch_bounds__(256)
void attn(const bf16_t* __restrict__ Q, const bf16_t* __restrict__ K,
          const bf16_t* __restrict__ Vt, const float* __restrict__ mask,
          float* __restrict__ out)
{
    __shared__ __align__(16) bf16_t Klds[32 * HDIM];     // [key][d]
    __shared__ __align__(16) bf16_t VldsT[HDIM * 32];    // [d][key]
    __shared__ __align__(16) bf16_t Plds[8][16 * 32];    // per-wave P tile

    const int tid  = threadIdx.x;
    const int lane = tid & 31;
    const int wave = tid >> 5;                 // 0..7
    const int h    = blockIdx.y;
    const int q0   = blockIdx.x * 128 + wave * 16;

    const int half = lane >> 4;
    const int lp   = lane & 15;

    const bf16_t* Qh  = Q  + (size_t)h * SEQ * HDIM;
    const bf16_t* Kh  = K  + (size_t)h * SEQ * HDIM;
    const bf16_t* Vth = Vt + (size_t)h * HDIM * SEQ;

    // Q fragments: two 16x32 A-tiles, each = two contiguous 16B runs per lane
    FragBF qa[2];
    {
        const bf16_t* qrow = Qh + (size_t)(q0 + lp) * HDIM;
#pragma unroll
        for (int c = 0; c < 2; ++c) {
            qa[c].q[0] = *(const uint4*)(qrow + c * 32 + (half << 3));
            qa[c].q[1] = *(const uint4*)(qrow + c * 32 + 16 + (half << 3));
        }
    }

    v8f o[4] = {};
    float rm[8], rs[8];
#pragma unroll
    for (int r = 0; r < 8; ++r) { rm[r] = -1e30f; rs[r] = 0.f; }

    const float scale = 0.125f;   // 1/sqrt(64)

    for (int key0 = 0; key0 < SEQ; key0 += 32) {
        __syncthreads();
#if defined(HAVE_ASYNC_LDS)
        {
            // K: 4KB contiguous -> one b128 per thread
            int e = tid * 8;                               // bf16 elements
            __builtin_amdgcn_global_load_async_to_lds_b128(
                AS1(Kh + (size_t)key0 * HDIM + e), AS3(Klds + e), 0, 0);
            // V: 64 rows x 64B -> one b128 per thread
            int row = tid >> 2;
            int c8  = (tid & 3) * 8;
            __builtin_amdgcn_global_load_async_to_lds_b128(
                AS1(Vth + (size_t)row * SEQ + key0 + c8),
                AS3(VldsT + row * 32 + c8), 0, 0);
        }
        __builtin_amdgcn_s_wait_asynccnt(0);
#else
        {
            ((uint4*)Klds)[tid] = ((const uint4*)(Kh + (size_t)key0 * HDIM))[tid];
            int row = tid >> 2;
            int c8  = (tid & 3) * 8;
            *(uint4*)(VldsT + row * 32 + c8) =
                *(const uint4*)(Vth + (size_t)row * SEQ + key0 + c8);
        }
#endif
        if (key0 + 32 < SEQ) {
            __builtin_prefetch(Kh + (size_t)(key0 + 32) * HDIM, 0, 3);
            __builtin_prefetch(Vth + key0 + 32, 0, 3);
        }
        __syncthreads();

        // S = Q @ K^T : B-fragment = 8 contiguous d at Klds[key][c*32+8*half], 8 more at +16
        v8f s[2] = {};
#pragma unroll
        for (int t = 0; t < 2; ++t) {
#pragma unroll
            for (int c = 0; c < 2; ++c) {
                FragBF b;
                const bf16_t* kr = Klds + (t * 16 + lp) * HDIM + c * 32 + (half << 3);
                b.q[0] = *(const uint4*)(kr);
                b.q[1] = *(const uint4*)(kr + 16);
                s[t] = __builtin_amdgcn_wmma_f32_16x16x32_bf16(false, qa[c].v, false, b.v,
                                                               (short)0, s[t], false, false);
            }
        }

        const float mk0 = mask[key0 + lp];
        const float mk1 = mask[key0 + 16 + lp];
#pragma unroll
        for (int r = 0; r < 8; ++r) {
            s[0][r] = s[0][r] * scale + mk0;
            s[1][r] = s[1][r] * scale + mk1;
        }

        // online softmax (rows in 16-lane halves, columns across lanes)
#pragma unroll
        for (int r = 0; r < 8; ++r) {
            float cm = fmaxf(s[0][r], s[1][r]);
#pragma unroll
            for (int off = 1; off < 16; off <<= 1)
                cm = fmaxf(cm, __shfl_xor(cm, off, 32));
            float nm    = fmaxf(rm[r], cm);
            float alpha = __expf(rm[r] - nm);
            float p0    = __expf(s[0][r] - nm);
            float p1    = __expf(s[1][r] - nm);
            float cs    = p0 + p1;
#pragma unroll
            for (int off = 1; off < 16; off <<= 1)
                cs += __shfl_xor(cs, off, 32);
            rs[r] = rs[r] * alpha + cs;
            rm[r] = nm;
#pragma unroll
            for (int j = 0; j < 4; ++j) o[j][r] *= alpha;
            s[0][r] = p0;
            s[1][r] = p1;
        }

        // P: D-layout -> row-major LDS -> A-fragment (two b128 loads)
        bf16_t* P = Plds[wave];
#pragma unroll
        for (int r = 0; r < 8; ++r) {
            int m = r + (half << 3);
            P[m * 32 + lp]      = f2bf(s[0][r]);
            P[m * 32 + 16 + lp] = f2bf(s[1][r]);
        }
        FragBF pa;
        {
            const bf16_t* prow = P + lp * 32;
            pa.q[0] = *(const uint4*)(prow + (half << 3));
            pa.q[1] = *(const uint4*)(prow + 16 + (half << 3));
        }

        // O += P @ V : B-fragment = 8 contiguous keys at VldsT[d][8*half], 8 more at +16
#pragma unroll
        for (int j = 0; j < 4; ++j) {
            FragBF b;
            const bf16_t* vr = VldsT + (j * 16 + lp) * 32 + (half << 3);
            b.q[0] = *(const uint4*)(vr);
            b.q[1] = *(const uint4*)(vr + 16);
            o[j] = __builtin_amdgcn_wmma_f32_16x16x32_bf16(false, pa.v, false, b.v,
                                                           (short)0, o[j], false, false);
        }
    }

    // epilogue: normalize, merge heads, fp32 out [L][H]
#pragma unroll
    for (int j = 0; j < 4; ++j)
#pragma unroll
        for (int r = 0; r < 8; ++r) {
            int m = q0 + r + (half << 3);
            int d = j * 16 + lp;
            out[(size_t)m * HID + h * HDIM + d] = o[j][r] / rs[r];
        }
}

// ---------------------------------------------------------------------------
extern "C" void kernel_launch(void* const* d_in, const int* in_sizes, int n_in,
                              void* d_out, int out_size, void* d_ws, size_t ws_size,
                              hipStream_t stream) {
    const float* xin[3] = { (const float*)d_in[0], (const float*)d_in[1],
                            (const float*)d_in[2] };
    const float* mask = (const float*)d_in[3];
    const float* win[3] = { (const float*)d_in[4], (const float*)d_in[6],
                            (const float*)d_in[8] };
    const float* bq = (const float*)d_in[5];
    const float* bk = (const float*)d_in[7];
    const float* bv = (const float*)d_in[9];

    bf16_t* ws = (bf16_t*)d_ws;
    const size_t xper = (size_t)SEQ * HID;        // 3,145,728
    const size_t wper = (size_t)HID * HID;        //   589,824
    const size_t oper = (size_t)NHEAD * SEQ * HDIM;

    bf16_t* Xb  = ws;                  // 3 * xper
    bf16_t* Wb  = Xb + 3 * xper;       // 3 * wper
    bf16_t* Qb  = Wb + 3 * wper;       // oper
    bf16_t* Kb  = Qb + oper;
    bf16_t* Vtb = Kb + oper;

    // one-shot fp32 -> bf16 conversion of activations and weights
    for (int i = 0; i < 3; ++i) {
        to_bf16<<<(int)(xper / 8 + 255) / 256, 256, 0, stream>>>(
            xin[i], Xb + (size_t)i * xper, (int)xper);
        to_bf16<<<(int)(wper / 8 + 255) / 256, 256, 0, stream>>>(
            win[i], Wb + (size_t)i * wper, (int)wper);
    }

    dim3 g1(HID / 64, SEQ / 64, 3), b1(32);
    qkv_proj<<<g1, b1, 0, stream>>>(Xb, Wb, bq, bk, bv, Qb, Kb, Vtb);

    dim3 g2(SEQ / 128, NHEAD), b2(256);
    attn<<<g2, b2, 0, stream>>>(Qb, Kb, Vtb, mask, (float*)d_out);
}
